// Qwen2VisionTransformerWrapper_87797721465080
// MI455X (gfx1250) — matrix operand order repaired and checked by reading, not compiled, not run on hardware
//
#include <hip/hip_runtime.h>
#include <cstdint>

// ---------------- constants ----------------
#define SS   2048
#define EE   1280
#define HH   16
#define DD   80
#define DP   96      // D padded to multiple of 32 for WMMA K
#define FF   5120
#define LL   8
#define QKVN 3840    // 3*H*D
#define OUTN 3584
#define MROW 512     // S*E / (4*E)

typedef __attribute__((ext_vector_type(16))) _Float16 v16h;
typedef __attribute__((ext_vector_type(8)))  _Float16 v8h;
typedef __attribute__((ext_vector_type(8)))  float    v8f;
typedef __attribute__((ext_vector_type(4)))  unsigned int u32x4;
typedef __attribute__((ext_vector_type(8)))  int i32x8;
typedef __attribute__((ext_vector_type(4)))  int i32x4;

// ---------------- WMMA helpers ----------------
__device__ inline v8f wmma16(v16h a, v16h b, v8f c) {
  return __builtin_amdgcn_wmma_f32_16x16x32_f16(false, a, false, b, (short)0, c, false, false);
}

// A-matrix 16x32 f16 fragment (row = lane%16; hi=lane/16 selects K subset)
__device__ inline v16h load_frag_a(const _Float16* row, int hi) {
  v8h x = *reinterpret_cast<const v8h*>(row + hi * 8);
  v8h y = *reinterpret_cast<const v8h*>(row + 16 + hi * 8);
  v16h r;
#pragma unroll
  for (int i = 0; i < 8; i++) { r[i] = x[i]; r[i + 8] = y[i]; }
  return r;
}

// B-matrix 32x16 f16 fragment stored NT (row = output col n, K contiguous)
__device__ inline v16h load_frag_b(const _Float16* row, int hi) {
  v8h x = *reinterpret_cast<const v8h*>(row + hi * 8);
  v8h y = *reinterpret_cast<const v8h*>(row + 16 + hi * 8);
  v16h r;
#pragma unroll
  for (int i = 0; i < 8; i++) { r[i] = x[i]; r[i + 8] = y[i]; }
  return r;
}

__device__ inline uint32_t lds_off_of(const void* p) {
  return (uint32_t)(uintptr_t)p;   // low 32 bits of flat LDS address = LDS byte offset
}

// ---------------- CDNA5 TDM (TENSORcnt) and async (ASYNCcnt) paths ----------------
#define HAVE_TDM __has_builtin(__builtin_amdgcn_tensor_load_to_lds)

#if HAVE_TDM
// 2D tile load: rows x cols (f16 elements), row stride = K elements, into LDS (contiguous)
__device__ inline void tdm_load_2d(uint32_t lds_off, const void* gptr,
                                   int K, int rows, int cols) {
  uint64_t ga = (uint64_t)gptr;
  u32x4 g0;
  g0[0] = 1u;                                              // count=1 (valid user D#)
  g0[1] = lds_off;                                         // lds_addr
  g0[2] = (uint32_t)ga;                                    // global_addr[31:0]
  g0[3] = (uint32_t)((ga >> 32) & 0x1ffffffu) | (2u << 30); // addr[56:32] | type=2
  i32x8 g1;
  const unsigned dim1 = 1u << 20;                          // tensor rows (tiles in-bounds)
  g1[0] = (int)(1u << 16);                                 // data_size=1 (2 bytes)
  g1[1] = (int)(((unsigned)K & 0xffffu) << 16);            // tensor_dim0 lo16
  g1[2] = (int)((((unsigned)K >> 16) & 0xffffu) | ((dim1 & 0xffffu) << 16));
  g1[3] = (int)(((dim1 >> 16) & 0xffffu) | ((unsigned)cols << 16)); // dim1 hi | tile_dim0
  g1[4] = (int)((unsigned)rows & 0xffffu);                 // tile_dim1 (tile_dim2=0)
  g1[5] = (int)(unsigned)K;                                // tensor_dim0_stride lo32
  g1[6] = 0;
  g1[7] = 0;
  i32x4 z4 = {0, 0, 0, 0};
#if __clang_major__ >= 23
  i32x8 z8 = {0, 0, 0, 0, 0, 0, 0, 0};
  __builtin_amdgcn_tensor_load_to_lds(g0, g1, z4, z4, z8, 0);
#else
  __builtin_amdgcn_tensor_load_to_lds(g0, g1, z4, z4, 0);
#endif
}
__device__ inline void wait_tensor_2() { __builtin_amdgcn_s_wait_tensorcnt(2); }
__device__ inline void wait_tensor_0() { __builtin_amdgcn_s_wait_tensorcnt(0); }
#else
__device__ inline void async_b128(uint32_t lds_off, uint64_t gaddr) {
  asm volatile("global_load_async_to_lds_b128 %0, %1, off"
               :: "v"(lds_off), "v"(gaddr) : "memory");
}
__device__ inline void wait_async_le6() { asm volatile("s_wait_asynccnt 0x6" ::: "memory"); }
__device__ inline void wait_async_0()   { asm volatile("s_wait_asynccnt 0x0" ::: "memory"); }
#endif

// ---------------- fp32 -> f16 convert ----------------
__global__ __launch_bounds__(256) void f32_to_f16(const float* __restrict__ src,
                                                  _Float16* __restrict__ dst, size_t n) {
  size_t i = (size_t)blockIdx.x * blockDim.x + threadIdx.x;
  size_t stride = (size_t)gridDim.x * blockDim.x;
  for (; i < n; i += stride) dst[i] = (_Float16)src[i];
}

// ---------------- LayerNorm (fp32 in, f16 out), one block per row ----------------
__global__ __launch_bounds__(256) void ln_f16(const float* __restrict__ x,
                                              const float* __restrict__ w,
                                              const float* __restrict__ b,
                                              _Float16* __restrict__ out, int E) {
  __shared__ float red[256];
  int row = blockIdx.x;
  const float* xr = x + (size_t)row * E;
  float s = 0.f, s2 = 0.f;
  for (int i = threadIdx.x; i < E; i += 256) { float v = xr[i]; s += v; s2 += v * v; }
  red[threadIdx.x] = s; __syncthreads();
  for (int off = 128; off > 0; off >>= 1) {
    if ((int)threadIdx.x < off) red[threadIdx.x] += red[threadIdx.x + off];
    __syncthreads();
  }
  float mean = red[0] / (float)E; __syncthreads();
  red[threadIdx.x] = s2; __syncthreads();
  for (int off = 128; off > 0; off >>= 1) {
    if ((int)threadIdx.x < off) red[threadIdx.x] += red[threadIdx.x + off];
    __syncthreads();
  }
  float var = red[0] / (float)E - mean * mean;
  float inv = rsqrtf(var + 1e-6f);
  for (int i = threadIdx.x; i < E; i += 256)
    out[(size_t)row * E + i] = (_Float16)((xr[i] - mean) * inv * w[i] + b[i]);
}

// ---------------- generic NT WMMA GEMM: C[M][N] = A[M][K] * B[N][K]^T + bias ----------------
// EP: 0 = store f32 ; 1 = residual add in-place into outF (fp32) ; 2 = quick-gelu -> f16
//     3 = exact-gelu -> f16 ; 4 = store f32 (final out)
template <int EP>
__device__ inline void store_frag(const v8f& c, int mrow0, int n, float bias_n,
                                  float* outF, _Float16* outH, int N) {
#pragma unroll
  for (int rr = 0; rr < 8; rr++) {
    int m = mrow0 + rr;
    float v = c[rr] + bias_n;
    size_t idx = (size_t)m * N + n;
    if constexpr (EP == 0) { outF[idx] = v; }
    else if constexpr (EP == 1) { outF[idx] = outF[idx] + v; }
    else if constexpr (EP == 2) {
      float g = v / (1.f + __expf(-1.702f * v));
      outH[idx] = (_Float16)g;
    } else if constexpr (EP == 3) {
      float g = 0.5f * v * (1.f + erff(v * 0.70710678f));
      outH[idx] = (_Float16)g;
    } else { outF[idx] = v; }
  }
}

// Block tile 128(M) x 256(N), K-slab 32, double-buffered LDS staging filled by the
// Tensor Data Mover (wave 0 issues one descriptor per tile; TENSORcnt tracks it).
// 8 waves = 2(M) x 4(N); wave tile 64x64 -> 16 WMMA per 16 ds_load_b128.
template <int EP>
__global__ __launch_bounds__(256) void gemm_nt(const _Float16* __restrict__ A,
                                               const _Float16* __restrict__ B,
                                               const float* __restrict__ bias,
                                               float* __restrict__ outF,
                                               _Float16* __restrict__ outH,
                                               int M, int N, int K) {
  __shared__ _Float16 As[2][128 * 32];
  __shared__ _Float16 Bs[2][256 * 32];

  int tid = threadIdx.x;
  int wave = tid >> 5, lane = tid & 31;
  int r = lane & 15, hi = lane >> 4;
  int mblock = blockIdx.y * 128, nblock = blockIdx.x * 256;
  int mw = (wave >> 2) * 64;   // wave M offset in block
  int nw = (wave & 3) * 64;    // wave N offset in block

  const _Float16* Ab = A + (size_t)mblock * K;
  const _Float16* Bb = B + (size_t)nblock * K;

#if !HAVE_TDM
  // fallback: per-lane async global->LDS. A: 512 chunks (2/thread); B: 1024 (4/thread)
  int rA = tid >> 1, cA = (tid & 1) * 16;
  const _Float16* gA = Ab + (size_t)rA * K + cA;
  const _Float16* gB = Bb + (size_t)tid * K;
  uint32_t lA[2], lB[2];
#pragma unroll
  for (int s = 0; s < 2; s++) {
    lA[s] = lds_off_of(&As[s][rA * 32 + cA]);
    lB[s] = lds_off_of(&Bs[s][tid * 32]);
  }
#endif

  auto issue = [&](int kt) {
    int st = kt & 1;
    int k0 = kt * 32;
#if HAVE_TDM
    if (wave == 0) {
      tdm_load_2d(lds_off_of(&As[st][0]), Ab + k0, K, 128, 32);
      tdm_load_2d(lds_off_of(&Bs[st][0]), Bb + k0, K, 256, 32);
    }
#else
    async_b128(lA[st],      (uint64_t)(gA + k0));
    async_b128(lA[st] + 16, (uint64_t)(gA + k0 + 8));
    async_b128(lB[st],      (uint64_t)(gB + k0));
    async_b128(lB[st] + 16, (uint64_t)(gB + k0 + 8));
    async_b128(lB[st] + 32, (uint64_t)(gB + k0 + 16));
    async_b128(lB[st] + 48, (uint64_t)(gB + k0 + 24));
#endif
  };

  v8f c[4][4];
#pragma unroll
  for (int mi = 0; mi < 4; mi++)
#pragma unroll
    for (int ni = 0; ni < 4; ni++)
      c[mi][ni] = (v8f){0.f,0.f,0.f,0.f,0.f,0.f,0.f,0.f};

  int nk = K >> 5;
  issue(0);
#pragma unroll 2
  for (int kt = 0; kt < nk; ++kt) {
    if (kt + 1 < nk) {
      issue(kt + 1);
#if HAVE_TDM
      if (wave == 0) wait_tensor_2();
#else
      wait_async_le6();
#endif
    } else {
#if HAVE_TDM
      if (wave == 0) wait_tensor_0();
#else
      wait_async_0();
#endif
    }
    __syncthreads();
    int st = kt & 1;
    const _Float16* as = &As[st][0];
    const _Float16* bs = &Bs[st][0];
    v16h a[4], b[4];
#pragma unroll
    for (int mi = 0; mi < 4; mi++) a[mi] = load_frag_a(as + (mw + mi * 16 + r) * 32, hi);
#pragma unroll
    for (int ni = 0; ni < 4; ni++) b[ni] = load_frag_b(bs + (nw + ni * 16 + r) * 32, hi);
#pragma unroll
    for (int mi = 0; mi < 4; mi++)
#pragma unroll
      for (int ni = 0; ni < 4; ni++)
        c[mi][ni] = wmma16(a[mi], b[ni], c[mi][ni]);
    __syncthreads();
  }

#pragma unroll
  for (int ni = 0; ni < 4; ni++) {
    int n = nblock + nw + ni * 16 + r;
    float bn = bias[n];
#pragma unroll
    for (int mi = 0; mi < 4; mi++)
      store_frag<EP>(c[mi][ni], mblock + mw + mi * 16 + hi * 8, n, bn, outF, outH, N);
  }
}

// ---------------- RoPE + Q/K/V packing ----------------
__global__ __launch_bounds__(256) void rope_pack(const float* __restrict__ qkv,
                                                 const float* __restrict__ cosb,
                                                 const float* __restrict__ sinb,
                                                 _Float16* __restrict__ Qh,
                                                 _Float16* __restrict__ Kh,
                                                 _Float16* __restrict__ Vt) {
  int s = blockIdx.x;
  for (int idx = threadIdx.x; idx < HH * DP; idx += 256) {
    int hd = idx / DP, d = idx % DP;
    float qv = 0.f, kv = 0.f;
    if (d < DD) {
      size_t qi = (size_t)s * QKVN + hd * DD + d;
      float cs = cosb[s * DD + d], sn = sinb[s * DD + d];
      float q = qkv[qi], k = qkv[qi + EE];
      float qr = (d < DD / 2) ? -qkv[qi + DD / 2] : qkv[qi - DD / 2];
      float kr = (d < DD / 2) ? -qkv[qi + EE + DD / 2] : qkv[qi + EE - DD / 2];
      qv = q * cs + qr * sn;
      kv = k * cs + kr * sn;
      Vt[(size_t)hd * DP * SS + (size_t)d * SS + s] = (_Float16)qkv[qi + 2 * EE];
    }
    size_t o = (size_t)hd * SS * DP + (size_t)s * DP + d;
    Qh[o] = (_Float16)qv;
    Kh[o] = (_Float16)kv;
  }
}

// ---------------- attention: one workgroup = (head, 16 query rows) ----------------
#define ATTN_SC_OFF   0
#define ATTN_SP_OFF   131072
#define ATTN_RED_OFF  196608
#define ATTN_PART_OFF 197632
#define ATTN_LDS      238592

__global__ __launch_bounds__(256) void attn_kernel(const _Float16* __restrict__ Qh,
                                                   const _Float16* __restrict__ Kh,
                                                   const _Float16* __restrict__ Vt,
                                                   const float* __restrict__ maskf,
                                                   _Float16* __restrict__ Oh,
                                                   float scale) {
  extern __shared__ char smem[];
  float*    sc   = (float*)(smem + ATTN_SC_OFF);
  _Float16* sp   = (_Float16*)(smem + ATTN_SP_OFF);
  float*    red  = (float*)(smem + ATTN_RED_OFF);
  float*    part = (float*)(smem + ATTN_PART_OFF);

  int h = blockIdx.y, m0 = blockIdx.x * 16;
  int wave = threadIdx.x >> 5, lane = threadIdx.x & 31;
  int r = lane & 15, hi = lane >> 4;

  const _Float16* qbase = Qh + (size_t)h * SS * DP;
  const _Float16* kbase = Kh + (size_t)h * SS * DP;

  // phase 1: scores tile (16 x 2048); hoist Q fragments (loop-invariant)
  const _Float16* arow = qbase + (size_t)(m0 + r) * DP;
  v16h aq[3];
#pragma unroll
  for (int kk = 0; kk < 3; kk++) aq[kk] = load_frag_a(arow + kk * 32, hi);

  for (int nt = wave; nt < SS / 16; nt += 8) {
    int n0 = nt * 16;
    const _Float16* brow = kbase + (size_t)(n0 + r) * DP;
    v8f c = {0.f,0.f,0.f,0.f,0.f,0.f,0.f,0.f};
#pragma unroll
    for (int kk = 0; kk < 3; kk++)
      c = wmma16(aq[kk], load_frag_b(brow + kk * 32, hi), c);
    int t = n0 + r;
#pragma unroll
    for (int rr = 0; rr < 8; rr++) {
      int m = rr + hi * 8;
      float v = c[rr] * scale +
                (1.0f - maskf[(size_t)(m0 + m) * SS + t]) * -3.4028235e38f;
      sc[m * SS + t] = v;
    }
  }
  __syncthreads();

  // phase 2: full-row softmax, emit f16 probs
  {
    int row = threadIdx.x >> 4;
    int j = threadIdx.x & 15;
    float lmax = -3.4e38f;
    for (int t = j * 128; t < j * 128 + 128; t++) lmax = fmaxf(lmax, sc[row * SS + t]);
    red[row * 16 + j] = lmax; __syncthreads();
    float rmax = -3.4e38f;
    for (int q = 0; q < 16; q++) rmax = fmaxf(rmax, red[row * 16 + q]);
    __syncthreads();
    float lsum = 0.f;
    for (int t = j * 128; t < j * 128 + 128; t++) {
      float e = __expf(sc[row * SS + t] - rmax);
      sc[row * SS + t] = e;
      lsum += e;
    }
    red[row * 16 + j] = lsum; __syncthreads();
    float rsum = 0.f;
    for (int q = 0; q < 16; q++) rsum += red[row * 16 + q];
    float rinv = 1.0f / rsum;
    for (int t = j * 128; t < j * 128 + 128; t++)
      sp[row * SS + t] = (_Float16)(sc[row * SS + t] * rinv);
  }
  __syncthreads();

  // phase 3: O = P x V ; each wave owns a 256-key K slice; 5 n-tiles cover d=0..79
  v8f acc[5];
#pragma unroll
  for (int i = 0; i < 5; i++) acc[i] = (v8f){0.f,0.f,0.f,0.f,0.f,0.f,0.f,0.f};
  const _Float16* vbase = Vt + (size_t)h * DP * SS;
  const _Float16* prow = sp + r * SS;
  for (int kk = 0; kk < 8; kk++) {
    int k0 = wave * 256 + kk * 32;
    v16h a = load_frag_a(prow + k0, hi);
#pragma unroll
    for (int nt = 0; nt < 5; nt++) {
      v16h b = load_frag_b(vbase + (size_t)(nt * 16 + r) * SS + k0, hi);
      acc[nt] = wmma16(a, b, acc[nt]);
    }
  }
#pragma unroll
  for (int nt = 0; nt < 5; nt++)
#pragma unroll
    for (int rr = 0; rr < 8; rr++)
      part[wave * 1280 + nt * 256 + (rr + hi * 8) * 16 + r] = acc[nt][rr];
  __syncthreads();
  for (int idx = threadIdx.x; idx < 1280; idx += 256) {
    float s = 0.f;
    for (int w2 = 0; w2 < 8; w2++) s += part[w2 * 1280 + idx];
    int nt = idx >> 8, within = idx & 255, m = within >> 4, d = nt * 16 + (within & 15);
    Oh[(size_t)(m0 + m) * EE + h * DD + d] = (_Float16)s;
  }
}

// ---------------- workspace layout (bytes) ----------------
#define OFF_QKVW  ((size_t)0)
#define OFF_PROJW ((size_t)78643200)
#define OFF_FC1W  ((size_t)104857600)
#define OFF_FC2W  ((size_t)209715200)
#define OFF_M1W   ((size_t)314572800)
#define OFF_M2W   ((size_t)367001600)
#define OFF_H     ((size_t)403701760)
#define OFF_X16   ((size_t)414187520)
#define OFF_QKV32 ((size_t)419430400)
#define OFF_QH    ((size_t)450887680)
#define OFF_KH    ((size_t)457179136)
#define OFF_VT    ((size_t)463470592)
#define OFF_O16   ((size_t)469762048)
#define OFF_Y1    ((size_t)475004928)
#define OFF_G1    ((size_t)495976448)

extern "C" void kernel_launch(void* const* d_in, const int* in_sizes, int n_in,
                              void* d_out, int out_size, void* d_ws, size_t ws_size,
                              hipStream_t stream) {
  const float* hs    = (const float*)d_in[0];
  const float* maskf = (const float*)d_in[1];
  const float* cosb  = (const float*)d_in[2];
  const float* sinb  = (const float*)d_in[3];
  const float* qkv_w = (const float*)d_in[4];
  const float* qkv_b = (const float*)d_in[5];
  const float* proj_w = (const float*)d_in[6];
  const float* proj_b = (const float*)d_in[7];
  const float* ln1_w = (const float*)d_in[8];
  const float* ln1_b = (const float*)d_in[9];
  const float* ln2_w = (const float*)d_in[10];
  const float* ln2_b = (const float*)d_in[11];
  const float* fc1_w = (const float*)d_in[12];
  const float* fc1_b = (const float*)d_in[13];
  const float* fc2_w = (const float*)d_in[14];
  const float* fc2_b = (const float*)d_in[15];
  const float* mln_w = (const float*)d_in[16];
  const float* mln_b = (const float*)d_in[17];
  const float* m1_w = (const float*)d_in[18];
  const float* m1_b = (const float*)d_in[19];
  const float* m2_w = (const float*)d_in[20];
  const float* m2_b = (const float*)d_in[21];

  char* ws = (char*)d_ws;
  _Float16* qkvw16 = (_Float16*)(ws + OFF_QKVW);
  _Float16* projw16 = (_Float16*)(ws + OFF_PROJW);
  _Float16* fc1w16 = (_Float16*)(ws + OFF_FC1W);
  _Float16* fc2w16 = (_Float16*)(ws + OFF_FC2W);
  _Float16* m1w16 = (_Float16*)(ws + OFF_M1W);
  _Float16* m2w16 = (_Float16*)(ws + OFF_M2W);
  float*    h     = (float*)(ws + OFF_H);
  _Float16* x16   = (_Float16*)(ws + OFF_X16);
  float*    qkv32 = (float*)(ws + OFF_QKV32);
  _Float16* Qh    = (_Float16*)(ws + OFF_QH);
  _Float16* Kh    = (_Float16*)(ws + OFF_KH);
  _Float16* Vt    = (_Float16*)(ws + OFF_VT);
  _Float16* o16   = (_Float16*)(ws + OFF_O16);
  _Float16* y1    = (_Float16*)(ws + OFF_Y1);
  _Float16* g1    = (_Float16*)(ws + OFF_G1);

  // one-time weight conversion to f16 (graph-safe: re-run every launch)
  f32_to_f16<<<2048, 256, 0, stream>>>(qkv_w, qkvw16, (size_t)LL * QKVN * EE);
  f32_to_f16<<<2048, 256, 0, stream>>>(proj_w, projw16, (size_t)LL * EE * EE);
  f32_to_f16<<<2048, 256, 0, stream>>>(fc1_w, fc1w16, (size_t)LL * FF * EE);
  f32_to_f16<<<2048, 256, 0, stream>>>(fc2_w, fc2w16, (size_t)LL * EE * FF);
  f32_to_f16<<<2048, 256, 0, stream>>>(m1_w, m1w16, (size_t)FF * 4 * EE);
  f32_to_f16<<<2048, 256, 0, stream>>>(m2_w, m2w16, (size_t)OUTN * FF);

  (void)hipMemcpyAsync(h, hs, (size_t)SS * EE * sizeof(float),
                       hipMemcpyDeviceToDevice, stream);

  (void)hipFuncSetAttribute((const void*)attn_kernel,
                            hipFuncAttributeMaxDynamicSharedMemorySize, ATTN_LDS);

  const float scale = 0.11180339887498949f; // 1/sqrt(80)

  for (int l = 0; l < LL; l++) {
    ln_f16<<<SS, 256, 0, stream>>>(h, ln1_w + l * EE, ln1_b + l * EE, x16, EE);
    gemm_nt<0><<<dim3(QKVN / 256, SS / 128), 256, 0, stream>>>(
        x16, qkvw16 + (size_t)l * QKVN * EE, qkv_b + (size_t)l * QKVN,
        qkv32, nullptr, SS, QKVN, EE);
    rope_pack<<<SS, 256, 0, stream>>>(qkv32, cosb, sinb, Qh, Kh, Vt);
    attn_kernel<<<dim3(SS / 16, HH), 256, ATTN_LDS, stream>>>(Qh, Kh, Vt, maskf, o16, scale);
    gemm_nt<1><<<dim3(EE / 256, SS / 128), 256, 0, stream>>>(
        o16, projw16 + (size_t)l * EE * EE, proj_b + (size_t)l * EE,
        h, nullptr, SS, EE, EE);
    ln_f16<<<SS, 256, 0, stream>>>(h, ln2_w + l * EE, ln2_b + l * EE, x16, EE);
    gemm_nt<2><<<dim3(FF / 256, SS / 128), 256, 0, stream>>>(
        x16, fc1w16 + (size_t)l * FF * EE, fc1_b + (size_t)l * FF,
        nullptr, y1, SS, FF, EE);
    gemm_nt<1><<<dim3(EE / 256, SS / 128), 256, 0, stream>>>(
        y1, fc2w16 + (size_t)l * EE * FF, fc2_b + (size_t)l * EE,
        h, nullptr, SS, EE, FF);
  }

  ln_f16<<<SS, 256, 0, stream>>>(h, mln_w, mln_b, x16, EE);
  gemm_nt<3><<<dim3(FF / 256, MROW / 128), 256, 0, stream>>>(
      x16, m1w16, m1_b, nullptr, g1, MROW, FF, 4 * EE);
  gemm_nt<4><<<dim3(OUTN / 256, MROW / 128), 256, 0, stream>>>(
      g1, m2w16, m2_b, (float*)d_out, nullptr, MROW, OUTN, FF);
}
